// CoAttentionNet_7541962572380
// MI455X (gfx1250) — matrix operand order repaired and checked by reading, not compiled, not run on hardware
//
#include <hip/hip_runtime.h>
#include <cstdint>

#define NN 4096
#define DD 64
#define OUTW 4224
#define SCALE 0.125f    // 1/sqrt(64)
#define NSPLIT 4
#define MBLK 32
#define ITERS (NN / NSPLIT / MBLK)   // 32

typedef __attribute__((ext_vector_type(16))) __bf16 v16bf;
typedef __attribute__((ext_vector_type(8)))  __bf16 v8bf;
typedef __attribute__((ext_vector_type(8)))  float  v8f;
typedef __attribute__((ext_vector_type(4)))  unsigned int u32x4;
typedef __attribute__((ext_vector_type(8)))  int i32x8;
typedef __attribute__((ext_vector_type(4)))  int i32x4;

#if defined(__has_builtin)
#if __has_builtin(__builtin_amdgcn_tensor_load_to_lds) && \
    __has_builtin(__builtin_amdgcn_s_wait_tensorcnt)
#define USE_TDM 1
#endif
#if __has_builtin(__builtin_amdgcn_permlane16)
#define USE_PERMLANE 1
#endif
#endif
#ifndef USE_TDM
#define USE_TDM 0
#endif
#ifndef USE_PERMLANE
#define USE_PERMLANE 0
#endif

static __device__ inline v8f wmma_bf16(v16bf a, v16bf b, v8f c) {
  return __builtin_amdgcn_wmma_f32_16x16x32_bf16(false, a, false, b, (short)0, c,
                                                 false, false);
}

static __device__ inline v8f vzero8() {
  v8f z;
#pragma unroll
  for (int i = 0; i < 8; ++i) z[i] = 0.0f;
  return z;
}

// ---- intra-16-lane butterfly reductions (rows live in one 16-lane half) ----
// v_permlane16_b32: VALU lane permute within each 16-lane row; sel table is
// 4 bits/lane in {SRC2,SRC1} (SRC1 = lanes 0-7). sel[i] = i ^ mask.
#if USE_PERMLANE
static __device__ inline float pl16(float x, int s0, int s1) {
  const int xi = __builtin_bit_cast(int, x);
  return __builtin_bit_cast(
      float, __builtin_amdgcn_permlane16(xi, xi, s0, s1, false, false));
}
static __device__ inline float rmax16(float x) {
  x = fmaxf(x, pl16(x, (int)0x67452301u, (int)0xEFCDAB89u));  // xor 1
  x = fmaxf(x, pl16(x, (int)0x54761032u, (int)0xDCFE98BAu));  // xor 2
  x = fmaxf(x, pl16(x, (int)0x32107654u, (int)0xBA98FEDCu));  // xor 4
  x = fmaxf(x, pl16(x, (int)0xFEDCBA98u, (int)0x76543210u));  // xor 8
  return x;
}
static __device__ inline float radd16(float x) {
  x += pl16(x, (int)0x67452301u, (int)0xEFCDAB89u);
  x += pl16(x, (int)0x54761032u, (int)0xDCFE98BAu);
  x += pl16(x, (int)0x32107654u, (int)0xBA98FEDCu);
  x += pl16(x, (int)0xFEDCBA98u, (int)0x76543210u);
  return x;
}
#else
static __device__ inline float rmax16(float x) {
#pragma unroll
  for (int msk = 1; msk < 16; msk <<= 1) x = fmaxf(x, __shfl_xor(x, msk, 32));
  return x;
}
static __device__ inline float radd16(float x) {
#pragma unroll
  for (int msk = 1; msk < 16; msk <<= 1) x += __shfl_xor(x, msk, 32);
  return x;
}
#endif

// Combine two 16B-aligned halves into a fragment register pair.
static __device__ inline v16bf load16x2(const __bf16* p) {
  v8bf lo = *(const v8bf*)p;
  v8bf hi = *(const v8bf*)(p + 8);
  v16bf r;
#pragma unroll
  for (int i = 0; i < 8; ++i) { r[i] = lo[i]; r[i + 8] = hi[i]; }
  return r;
}

// A-fragment loader: 16x32 bf16 tile, row-major source (stride in elements).
// Layout: lane -> M=lane%16, K in {k0+g*16..+15} (contiguous, matches B).
static __device__ inline v16bf load_a_bf16(const __bf16* base, int stride,
                                           int row0, int k0, int ln, int g) {
  const __bf16* p = base + (size_t)(row0 + ln) * stride + k0 + g * 16;
  v8bf lo = *(const v8bf*)p;
  v8bf hi = *(const v8bf*)(p + 8);
  v16bf a;
#pragma unroll
  for (int i = 0; i < 8; ++i) { a[i] = lo[i]; a[i + 8] = hi[i]; }
  return a;
}

static __device__ inline uint32_t lds_off(const void* p) {
  // LDS aperture: addr[31:0] is the LDS byte offset (ISA 10.2).
  return (uint32_t)(uintptr_t)p;
}

#if USE_TDM
// 2-D TDM load, bf16 elements, with LDS row padding for bank-conflict-free
// reads. Descriptor bitfields per CDNA5 ISA §8.3-8.6.
static __device__ inline void tdm_load_2d(uint32_t lds_byte, const void* gptr,
                                          uint32_t td0, uint32_t td1,
                                          uint32_t tile0, uint32_t tile1,
                                          uint32_t stride0,
                                          uint32_t pad_int, uint32_t pad_amt) {
  const uint64_t ga = (uint64_t)gptr;
  u32x4 g0;
  g0[0] = 1u;                                   // count=1, user descriptor
  g0[1] = lds_byte;                             // lds_addr
  g0[2] = (uint32_t)ga;                         // global_addr[31:0]
  g0[3] = (uint32_t)((ga >> 32) & 0x1FFFFFFu) | (2u << 30);  // addr[56:32],type=2
  i32x8 g1;
  g1[0] = (int)((1u << 16) |                    // data_size = 2 bytes
                (1u << 20) |                    // pad_enable
                (pad_int << 22) | (pad_amt << 25));
  g1[1] = (int)((td0 & 0xFFFFu) << 16);
  g1[2] = (int)(((td0 >> 16) & 0xFFFFu) | ((td1 & 0xFFFFu) << 16));
  g1[3] = (int)(((td1 >> 16) & 0xFFFFu) | (tile0 << 16));
  g1[4] = (int)tile1;
  g1[5] = (int)stride0;
  g1[6] = 0;
  g1[7] = 0;
  const i32x4 z4 = {0, 0, 0, 0};
  const i32x8 z8 = {0, 0, 0, 0, 0, 0, 0, 0};
  // 6-arg variant (clang-23 / therock-10.0): groups 2/3 + extra group unused
  // for a 2-D tensor; cpol = 0 (RT).
  __builtin_amdgcn_tensor_load_to_lds(g0, g1, z4, z4, z8, 0);
}
#endif

// ---------------- Phase 1: WMMA projections (X@W + b -> bf16) ----------------
template <int TR>
__global__ __launch_bounds__(32) void proj_wmma(const float* __restrict__ X,
                                                const float* __restrict__ W,
                                                const float* __restrict__ Bv,
                                                __bf16* __restrict__ O) {
  const int nbase = blockIdx.x * 16;
  const int lane = threadIdx.x & 31;
  const int ln = lane & 15, g = lane >> 4;

  v16bf aX[2];
#pragma unroll
  for (int kk = 0; kk < 2; ++kk) {
    const float* xr = X + (size_t)(nbase + ln) * DD + kk * 32 + g * 16;
    v16bf a;
#pragma unroll
    for (int i = 0; i < 8; ++i) { a[i] = (__bf16)xr[i]; a[i + 8] = (__bf16)xr[8 + i]; }
    aX[kk] = a;
  }

#pragma unroll
  for (int cg = 0; cg < 4; ++cg) {
    v8f acc = vzero8();
#pragma unroll
    for (int kk = 0; kk < 2; ++kk) {
      v16bf b;  // B[k][n]: e -> k = kk*32 + g*16 + e, n = cg*16+ln
#pragma unroll
      for (int e = 0; e < 16; ++e)
        b[e] = (__bf16)W[(size_t)(kk * 32 + g * 16 + e) * DD + cg * 16 + ln];
      acc = wmma_bf16(aX[kk], b, acc);
    }
    const float bb = Bv[cg * 16 + ln];
#pragma unroll
    for (int v = 0; v < 8; ++v) {  // C layout: row = v + 8*g, col = cg*16+ln
      const float val = acc[v] + bb;
      const int row = nbase + v + 8 * g;
      const int col = cg * 16 + ln;
      if (TR) O[(size_t)col * NN + row] = (__bf16)val;
      else    O[(size_t)row * DD + col] = (__bf16)val;
    }
  }
}

// -------- Phase 2: fused dual flash-attention + coupling, m-split 4x --------
__global__ __launch_bounds__(128) void coattn_flash(
    const __bf16* __restrict__ Qi, const __bf16* __restrict__ Ki,
    const __bf16* __restrict__ VtI,
    const __bf16* __restrict__ Qm, const __bf16* __restrict__ Km,
    const __bf16* __restrict__ VtM,
    float* __restrict__ Stat, float* __restrict__ Part) {
  // TDM-staged operand tiles, double buffered; padded rows (144B / 80B) so
  // B-fragment ds reads are spread across banks.
  __shared__ __bf16 KiT[2][MBLK][72];
  __shared__ __bf16 KmT[2][MBLK][72];
  __shared__ __bf16 ViT[2][DD][40];
  __shared__ __bf16 VmT[2][DD][40];
  __shared__ __bf16 Pbuf[4][2][16][40];

  const int wave = threadIdx.x >> 5;
  const int lane = threadIdx.x & 31;
  const int tid = threadIdx.x;
  const int ln = lane & 15, g = (lane >> 4) & 1;
  const int nbase = (blockIdx.x * 4 + wave) * 16;
  const int split = blockIdx.y;
  const int m0 = split * (NN / NSPLIT);

  v16bf aQm[2], aQi[2];
#pragma unroll
  for (int kk = 0; kk < 2; ++kk) {
    aQm[kk] = load_a_bf16(Qm, DD, nbase, kk * 32, ln, g);
    aQi[kk] = load_a_bf16(Qi, DD, nbase, kk * 32, ln, g);
  }

  v8f Oim[4], Omi[4];
#pragma unroll
  for (int cg = 0; cg < 4; ++cg) { Oim[cg] = vzero8(); Omi[cg] = vzero8(); }

  float m_im[8], l_im[8], m_mi[8], l_mi[8], c_acc[8];
#pragma unroll
  for (int v = 0; v < 8; ++v) {
    m_im[v] = -__builtin_inff(); m_mi[v] = -__builtin_inff();
    l_im[v] = 0.f; l_mi[v] = 0.f; c_acc[v] = 0.f;
  }

  for (int it = 0; it < ITERS; ++it) {
    const int mb = m0 + it * MBLK;
#if USE_TDM
    const int cur = it & 1;
    if (it == 0) {
      if (wave == 0) {
        tdm_load_2d(lds_off(&KiT[0][0][0]), Ki + (size_t)mb * DD, DD, NN, 64, 32, DD, 4, 3);
        tdm_load_2d(lds_off(&KmT[0][0][0]), Km + (size_t)mb * DD, DD, NN, 64, 32, DD, 4, 3);
        tdm_load_2d(lds_off(&ViT[0][0][0]), VtI + mb, NN, DD, 32, 64, NN, 3, 3);
        tdm_load_2d(lds_off(&VmT[0][0][0]), VtM + mb, NN, DD, 32, 64, NN, 3, 3);
        __builtin_amdgcn_s_wait_tensorcnt(0);
      }
      __syncthreads();
    }
    if (wave == 0 && it + 1 < ITERS) {  // prefetch next tile into other buffer
      const int nb = cur ^ 1, mn = mb + MBLK;
      tdm_load_2d(lds_off(&KiT[nb][0][0]), Ki + (size_t)mn * DD, DD, NN, 64, 32, DD, 4, 3);
      tdm_load_2d(lds_off(&KmT[nb][0][0]), Km + (size_t)mn * DD, DD, NN, 64, 32, DD, 4, 3);
      tdm_load_2d(lds_off(&ViT[nb][0][0]), VtI + mn, NN, DD, 32, 64, NN, 3, 3);
      tdm_load_2d(lds_off(&VmT[nb][0][0]), VtM + mn, NN, DD, 32, 64, NN, 3, 3);
    }
#else
    const int cur = 0;
    // Cooperative staging fallback (same padded layout).
    for (int idx = tid; idx < 256; idx += 128) {
      const int row = idx >> 3, ch = idx & 7;
      *(v8bf*)&KiT[0][row][ch * 8] = *(const v8bf*)(Ki + (size_t)(mb + row) * DD + ch * 8);
      *(v8bf*)&KmT[0][row][ch * 8] = *(const v8bf*)(Km + (size_t)(mb + row) * DD + ch * 8);
    }
    for (int idx = tid; idx < 256; idx += 128) {
      const int row = idx >> 2, ch = idx & 3;
      *(v8bf*)&ViT[0][row][ch * 8] = *(const v8bf*)(VtI + (size_t)row * NN + mb + ch * 8);
      *(v8bf*)&VmT[0][row][ch * 8] = *(const v8bf*)(VtM + (size_t)row * NN + mb + ch * 8);
    }
    __syncthreads();
#endif

    // ---- S tiles: S_IM = Qm*Ki^T, S_MI = Qi*Km^T ----
    v8f Sim[2], Smi[2];
#pragma unroll
    for (int cg = 0; cg < 2; ++cg) {
      v8f a1 = vzero8(), a2 = vzero8();
#pragma unroll
      for (int kk = 0; kk < 2; ++kk) {
        const v16bf bki = load16x2(&KiT[cur][cg * 16 + ln][kk * 32 + g * 16]);
        const v16bf bkm = load16x2(&KmT[cur][cg * 16 + ln][kk * 32 + g * 16]);
        a1 = wmma_bf16(aQm[kk], bki, a1);
        a2 = wmma_bf16(aQi[kk], bkm, a2);
      }
      Sim[cg] = a1; Smi[cg] = a2;
    }
#pragma unroll
    for (int v = 0; v < 8; ++v) {
      Sim[0][v] *= SCALE; Sim[1][v] *= SCALE;
      Smi[0][v] *= SCALE; Smi[1][v] *= SCALE;
    }

    // ---- online softmax: row max via permlane16 butterfly (no LDS) ----
    float tmax_im[8], tmax_mi[8];
#pragma unroll
    for (int v = 0; v < 8; ++v) {
      tmax_im[v] = rmax16(fmaxf(Sim[0][v], Sim[1][v]));
      tmax_mi[v] = rmax16(fmaxf(Smi[0][v], Smi[1][v]));
    }

    float r_im[8], r_mi[8];
#pragma unroll
    for (int v = 0; v < 8; ++v) {
      float nm = fmaxf(m_im[v], tmax_im[v]);
      r_im[v] = __expf(m_im[v] - nm); m_im[v] = nm;
      nm = fmaxf(m_mi[v], tmax_mi[v]);
      r_mi[v] = __expf(m_mi[v] - nm); m_mi[v] = nm;
    }
    float Pim0[8], Pim1[8], Pmi0[8], Pmi1[8];
#pragma unroll
    for (int v = 0; v < 8; ++v) {
      Pim0[v] = __expf(Sim[0][v] - m_im[v]);
      Pim1[v] = __expf(Sim[1][v] - m_im[v]);
      Pmi0[v] = __expf(Smi[0][v] - m_mi[v]);
      Pmi1[v] = __expf(Smi[1][v] - m_mi[v]);
      l_im[v] = l_im[v] * r_im[v] + Pim0[v] + Pim1[v];
      l_mi[v] = l_mi[v] * r_mi[v] + Pmi0[v] + Pmi1[v];
      c_acc[v] = c_acc[v] * r_im[v] * r_mi[v]
               + Pim0[v] * Pmi0[v] + Pim1[v] * Pmi1[v];
    }
#pragma unroll
    for (int cg = 0; cg < 4; ++cg)
#pragma unroll
      for (int v = 0; v < 8; ++v) { Oim[cg][v] *= r_im[v]; Omi[cg][v] *= r_mi[v]; }

    // ---- stage P (C-layout) to wave-private LDS, read back as A-frag ----
    // (wave-private buffer + in-order LDS per wave: no barrier needed)
#pragma unroll
    for (int v = 0; v < 8; ++v) {
      Pbuf[wave][0][v + 8 * g][ln]      = (__bf16)Pim0[v];
      Pbuf[wave][0][v + 8 * g][16 + ln] = (__bf16)Pim1[v];
      Pbuf[wave][1][v + 8 * g][ln]      = (__bf16)Pmi0[v];
      Pbuf[wave][1][v + 8 * g][16 + ln] = (__bf16)Pmi1[v];
    }
    v16bf aPim, aPmi;
    {
      const __bf16* p0 = &Pbuf[wave][0][ln][g * 16];
      const __bf16* p1 = &Pbuf[wave][1][ln][g * 16];
      v8bf lo0 = *(const v8bf*)p0, hi0 = *(const v8bf*)(p0 + 8);
      v8bf lo1 = *(const v8bf*)p1, hi1 = *(const v8bf*)(p1 + 8);
#pragma unroll
      for (int i = 0; i < 8; ++i) {
        aPim[i] = lo0[i]; aPim[i + 8] = hi0[i];
        aPmi[i] = lo1[i]; aPmi[i + 8] = hi1[i];
      }
    }

    // ---- O += P @ V ----
#pragma unroll
    for (int cg = 0; cg < 4; ++cg) {
      const v16bf bvi = load16x2(&ViT[cur][cg * 16 + ln][g * 16]);
      const v16bf bvm = load16x2(&VmT[cur][cg * 16 + ln][g * 16]);
      Oim[cg] = wmma_bf16(aPim, bvi, Oim[cg]);
      Omi[cg] = wmma_bf16(aPmi, bvm, Omi[cg]);
    }

#if USE_TDM
    if (wave == 0) __builtin_amdgcn_s_wait_tensorcnt(0);
    __syncthreads();
#else
    __syncthreads();
#endif
  }

  // ---- reduce row-wise partials across the 16-lane half (permlane adds) ----
#pragma unroll
  for (int v = 0; v < 8; ++v) {
    l_im[v]  = radd16(l_im[v]);
    l_mi[v]  = radd16(l_mi[v]);
    c_acc[v] = radd16(c_acc[v]);
  }

  // ---- emit per-split partial state ----
#pragma unroll
  for (int cg = 0; cg < 4; ++cg)
#pragma unroll
    for (int v = 0; v < 8; ++v) {
      const size_t row = (size_t)(nbase + v + 8 * g);
      float* pr = Part + ((size_t)split * NN + row) * 128;
      pr[cg * 16 + ln]      = Oim[cg][v];
      pr[64 + cg * 16 + ln] = Omi[cg][v];
    }
  if (ln == 0) {
#pragma unroll
    for (int v = 0; v < 8; ++v) {
      float* st = Stat + ((size_t)split * NN + (nbase + v + 8 * g)) * 8;
      st[0] = m_im[v]; st[1] = l_im[v]; st[2] = m_mi[v]; st[3] = l_mi[v];
      st[4] = c_acc[v];
    }
  }
}

// ---------------- Phase 2b: merge the 4 m-splits exactly ----------------
__global__ __launch_bounds__(64) void merge_splits(
    const float* __restrict__ Stat, const float* __restrict__ Part,
    float* __restrict__ out, float* __restrict__ c_out) {
  const int row = blockIdx.x;
  const int t = threadIdx.x;
  float mim[NSPLIT], lim[NSPLIT], mmi[NSPLIT], lmi[NSPLIT], cc[NSPLIT];
  float Mim = -__builtin_inff(), Mmi = -__builtin_inff();
#pragma unroll
  for (int s = 0; s < NSPLIT; ++s) {
    const float* st = Stat + ((size_t)s * NN + row) * 8;
    mim[s] = st[0]; lim[s] = st[1]; mmi[s] = st[2]; lmi[s] = st[3]; cc[s] = st[4];
    Mim = fmaxf(Mim, mim[s]); Mmi = fmaxf(Mmi, mmi[s]);
  }
  float Lim = 0.f, Lmi = 0.f, csum = 0.f, fim = 0.f, fmi = 0.f;
#pragma unroll
  for (int s = 0; s < NSPLIT; ++s) {
    const float aim = __expf(mim[s] - Mim);
    const float ami = __expf(mmi[s] - Mmi);
    Lim += aim * lim[s];
    Lmi += ami * lmi[s];
    csum += cc[s] * aim * ami;
    const float* pr = Part + ((size_t)s * NN + row) * 128;
    fim += aim * pr[t];
    fmi += ami * pr[64 + t];
  }
  out[(size_t)row * OUTW + t]      = fim / Lim;
  out[(size_t)row * OUTW + 64 + t] = fmi / Lmi;
  if (t == 0) c_out[row] = csum / (Lim * Lmi);
}

// ---------- Phase 3: A_final[n,d,e] = c[n]*X_M[n,d]*X_I[n,e] (BW-bound) ------
__global__ __launch_bounds__(256) void outer_prod(
    const float* __restrict__ Xi, const float* __restrict__ Xm,
    const float* __restrict__ c, float* __restrict__ out) {
  const int n = blockIdx.x;
  __shared__ float sXi[64];
  __shared__ float sCXm[64];
  const int t = threadIdx.x;
  if (t < 64) {
    sXi[t]  = Xi[(size_t)n * DD + t];
    sCXm[t] = c[n] * Xm[(size_t)n * DD + t];
  }
  __syncthreads();
  float* row = out + (size_t)n * OUTW + 128;
#pragma unroll
  for (int i = t; i < 1024; i += 256) {
    const int d = i >> 4;
    const int e4 = (i & 15) << 2;
    const float cm = sCXm[d];
    const float4 x = *(const float4*)(&sXi[e4]);
    float4 o; o.x = cm * x.x; o.y = cm * x.y; o.z = cm * x.z; o.w = cm * x.w;
    *(float4*)(row + d * DD + e4) = o;
  }
}

// ---------------------------------------------------------------------------
extern "C" void kernel_launch(void* const* d_in, const int* in_sizes, int n_in,
                              void* d_out, int out_size, void* d_ws, size_t ws_size,
                              hipStream_t stream) {
  const float* X_I = (const float*)d_in[0];
  const float* X_M = (const float*)d_in[1];
  const float* WqI = (const float*)d_in[2];  const float* bqI = (const float*)d_in[3];
  const float* WkI = (const float*)d_in[4];  const float* bkI = (const float*)d_in[5];
  const float* WvI = (const float*)d_in[6];  const float* bvI = (const float*)d_in[7];
  const float* WqM = (const float*)d_in[8];  const float* bqM = (const float*)d_in[9];
  const float* WkM = (const float*)d_in[10]; const float* bkM = (const float*)d_in[11];
  const float* WvM = (const float*)d_in[12]; const float* bvM = (const float*)d_in[13];

  const size_t matE = (size_t)NN * DD;
  __bf16* ws = (__bf16*)d_ws;
  __bf16* Qi  = ws + 0 * matE;
  __bf16* Ki  = ws + 1 * matE;
  __bf16* VtI = ws + 2 * matE;   // [64][4096]
  __bf16* Qm  = ws + 3 * matE;
  __bf16* Km  = ws + 4 * matE;
  __bf16* VtM = ws + 5 * matE;   // [64][4096]
  float* fws  = (float*)(ws + 6 * matE);
  float* c_ws = fws;                                   // [4096]
  float* Stat = fws + NN;                              // [4][4096][8]
  float* Part = Stat + (size_t)NSPLIT * NN * 8;        // [4][4096][128]

  proj_wmma<0><<<dim3(NN / 16), 32, 0, stream>>>(X_I, WqI, bqI, Qi);
  proj_wmma<0><<<dim3(NN / 16), 32, 0, stream>>>(X_I, WkI, bkI, Ki);
  proj_wmma<1><<<dim3(NN / 16), 32, 0, stream>>>(X_I, WvI, bvI, VtI);
  proj_wmma<0><<<dim3(NN / 16), 32, 0, stream>>>(X_M, WqM, bqM, Qm);
  proj_wmma<0><<<dim3(NN / 16), 32, 0, stream>>>(X_M, WkM, bkM, Km);
  proj_wmma<1><<<dim3(NN / 16), 32, 0, stream>>>(X_M, WvM, bvM, VtM);

  coattn_flash<<<dim3(NN / 64, NSPLIT), 128, 0, stream>>>(Qi, Ki, VtI, Qm, Km, VtM,
                                                          Stat, Part);
  merge_splits<<<dim3(NN), 64, 0, stream>>>(Stat, Part, (float*)d_out, c_ws);
  outer_prod<<<dim3(NN), 256, 0, stream>>>(X_I, X_M, c_ws, (float*)d_out);
}
